// Attention_62766652064032
// MI455X (gfx1250) — matrix operand order repaired and checked by reading, not compiled, not run on hardware
//
#include <hip/hip_runtime.h>

// ---------------------------------------------------------------------------
// Causal multi-head attention forward for MI455X (gfx1250), wave32 + WMMA.
//   B=2, S=2048, DIM=1024, H=16, HD=64
// Pipeline: f32->f16 convert -> 3x GEMM (QKV, f16 out) -> flash-attention
//           (online softmax, no SxS materialization) -> GEMM (out-proj, f32).
// Matrix ops on v_wmma_f32_16x16x32_f16; tile staging via gfx1250
// global_load_async_to_lds_b128 + s_wait_asynccnt.
// ---------------------------------------------------------------------------

#define DIMD   1024
#define NHEAD  16
#define HD     64
#define BATCH  2
#define SEQ    2048
#define MTOT   (BATCH * SEQ)   // 4096

typedef _Float16 h8  __attribute__((ext_vector_type(8)));
typedef _Float16 h16 __attribute__((ext_vector_type(16)));
typedef float    f8  __attribute__((ext_vector_type(8)));
typedef int      v4i __attribute__((ext_vector_type(4)));

union H16 { h16 v; h8 h[2]; };

#if defined(__has_builtin)
#if __has_builtin(__builtin_amdgcn_global_load_async_to_lds_b128)
#define HAVE_ASYNC_LDS 1
#endif
#endif
#ifndef HAVE_ASYNC_LDS
#define HAVE_ASYNC_LDS 0
#endif

template <int N>
__device__ __forceinline__ void wait_asynccnt()
{
#if defined(__has_builtin) && __has_builtin(__builtin_amdgcn_s_wait_asynccnt)
    __builtin_amdgcn_s_wait_asynccnt(N);
#else
    asm volatile("s_wait_asynccnt %0" ::"i"(N) : "memory");
#endif
}

#if HAVE_ASYNC_LDS
// One 16-byte global -> LDS async transfer per lane (no VGPR data path).
// Builtin signature (from compiler diagnostic): v4i addrspace-qualified ptrs.
__device__ __forceinline__ void async_copy_b128(const _Float16* g, _Float16* l)
{
    typedef __attribute__((address_space(1))) v4i* gp_t;  // global src
    typedef __attribute__((address_space(3))) v4i* lp_t;  // LDS dst
    __builtin_amdgcn_global_load_async_to_lds_b128(
        (gp_t)(v4i*)(void*)const_cast<_Float16*>(g),
        (lp_t)(v4i*)(void*)l,
        0, 0);
}
#endif

// ---------------------------------------------------------------------------
// fp32 -> f16 conversion (all sizes here are multiples of 4)
// ---------------------------------------------------------------------------
__global__ __launch_bounds__(256)
void cvt_f32_to_f16(const float* __restrict__ src, _Float16* __restrict__ dst, int n)
{
    int i = (blockIdx.x * 256 + threadIdx.x) * 4;
    if (i + 3 < n) {
        float4 f = *(const float4*)(src + i);
        dst[i + 0] = (_Float16)f.x;
        dst[i + 1] = (_Float16)f.y;
        dst[i + 2] = (_Float16)f.z;
        dst[i + 3] = (_Float16)f.w;
    }
}

// ---------------------------------------------------------------------------
// C[M,N] = A[M,K] @ W[N,K]^T   (both operands row-major along K)
// 128x128 block tile, 256 threads = 8 waves (2 x 4), each wave 64x32 out.
// Double-buffered LDS tiles, async global->LDS staging.
// ---------------------------------------------------------------------------
template <typename OutT>
__global__ __launch_bounds__(256)
void gemm_xwt(const _Float16* __restrict__ A, const _Float16* __restrict__ W,
              OutT* __restrict__ C, int M, int N, int K)
{
    __shared__ __align__(16) _Float16 lA[2][128 * 32];
    __shared__ __align__(16) _Float16 lB[2][128 * 32];

    const int tid  = threadIdx.x;
    const int wave = tid >> 5;          // 0..7
    const int lane = tid & 31;
    const int wm   = wave >> 2;         // 0..1 : M direction
    const int wn   = wave & 3;          // 0..3 : N direction
    const int m0   = blockIdx.y * 128;
    const int n0   = blockIdx.x * 128;

    const int lh = lane & 15;           // 0..15
    const int hi = lane >> 4;           // 0/1

    const int lrow = tid >> 2;          // 0..63 (staging row)
    const int lcol = (tid & 3) * 8;     // staging half-offset within 32

    f8 acc[4][2] = {};

    // Stage one 128x32 A tile + 128x32 B tile: 4 x b128 per thread.
    auto issue_tile = [&](int buf, int kk) {
#if HAVE_ASYNC_LDS
        async_copy_b128(&A[(size_t)(m0 + lrow     ) * K + kk + lcol], &lA[buf][(lrow     ) * 32 + lcol]);
        async_copy_b128(&A[(size_t)(m0 + lrow + 64) * K + kk + lcol], &lA[buf][(lrow + 64) * 32 + lcol]);
        async_copy_b128(&W[(size_t)(n0 + lrow     ) * K + kk + lcol], &lB[buf][(lrow     ) * 32 + lcol]);
        async_copy_b128(&W[(size_t)(n0 + lrow + 64) * K + kk + lcol], &lB[buf][(lrow + 64) * 32 + lcol]);
#else
        *(h8*)&lA[buf][(lrow     ) * 32 + lcol] = *(const h8*)&A[(size_t)(m0 + lrow     ) * K + kk + lcol];
        *(h8*)&lA[buf][(lrow + 64) * 32 + lcol] = *(const h8*)&A[(size_t)(m0 + lrow + 64) * K + kk + lcol];
        *(h8*)&lB[buf][(lrow     ) * 32 + lcol] = *(const h8*)&W[(size_t)(n0 + lrow     ) * K + kk + lcol];
        *(h8*)&lB[buf][(lrow + 64) * 32 + lcol] = *(const h8*)&W[(size_t)(n0 + lrow + 64) * K + kk + lcol];
#endif
    };

    issue_tile(0, 0);

    for (int kk = 0; kk < K; kk += 32) {
        const int cur = (kk >> 5) & 1;
        const bool more = (kk + 32) < K;
        if (more) issue_tile(cur ^ 1, kk + 32);  // prefetch next tile into other buffer

        // In-order async completion: <=4 outstanding means current tile landed.
        if (more) wait_asynccnt<4>(); else wait_asynccnt<0>();
        __syncthreads();

        // B fragments (32x16, KxN): lane holds col n=lh, halves k = hi*8 + [0..8) and hi*8 + 16 + [0..8)
        H16 bf[2];
#pragma unroll
        for (int jn = 0; jn < 2; ++jn) {
            const _Float16* bp = &lB[cur][(wn * 32 + jn * 16 + lh) * 32 + hi * 8];
            bf[jn].h[0] = *(const h8*)bp;
            bf[jn].h[1] = *(const h8*)(bp + 16);
        }
        // A fragments (16x32, MxK): lane holds row m=lh,
        // halves k = hi*8 + [0..8) and hi*8 + 16 + [0..8)
#pragma unroll
        for (int im = 0; im < 4; ++im) {
            const _Float16* ap = &lA[cur][(wm * 64 + im * 16 + lh) * 32 + hi * 8];
            H16 af;
            af.h[0] = *(const h8*)ap;
            af.h[1] = *(const h8*)(ap + 16);
#pragma unroll
            for (int jn = 0; jn < 2; ++jn) {
                acc[im][jn] = __builtin_amdgcn_wmma_f32_16x16x32_f16(
                    false, af.v, false, bf[jn].v, (short)0, acc[im][jn], false, false);
            }
        }
        __syncthreads();  // all readers done before this buffer is refilled
    }

    // C/D layout: lane holds col n=lh, rows m = j + hi*8
#pragma unroll
    for (int im = 0; im < 4; ++im)
#pragma unroll
        for (int jn = 0; jn < 2; ++jn)
#pragma unroll
            for (int j = 0; j < 8; ++j) {
                const int m = m0 + wm * 64 + im * 16 + j + hi * 8;
                const int n = n0 + wn * 32 + jn * 16 + lh;
                C[(size_t)m * N + n] = (OutT)acc[im][jn][j];
            }
}

// ---------------------------------------------------------------------------
// Flash attention: one wave per (b, h, 16-query block); 32 keys per iter.
// scores = Q Kt / 8 with causal mask, online softmax, O += P @ V.
// V block staged to LDS asynchronously, overlapping the softmax VALU work.
// ---------------------------------------------------------------------------
__global__ __launch_bounds__(128)
void flash_attn(const _Float16* __restrict__ Qm, const _Float16* __restrict__ Km,
                const _Float16* __restrict__ Vm, _Float16* __restrict__ Om)
{
    __shared__ __align__(16) _Float16 lV[4][32 * 64];  // per-wave V block
    __shared__ __align__(16) _Float16 lP[4][16 * 32];  // per-wave P transpose buf

    const int tid  = threadIdx.x;
    const int wv   = tid >> 5;
    const int lane = tid & 31;
    const int lh   = lane & 15;
    const int hi   = lane >> 4;

    const int task = blockIdx.x * 4 + wv;  // 0..4095
    const int qb   = task & 127;           // query block (16 queries)
    const int h    = (task >> 7) & 15;
    const int b    = task >> 11;

    const size_t rowstride = DIMD;  // halves
    const size_t base = (size_t)b * SEQ * DIMD + (size_t)h * HD;

    // Q A-fragments (16x32 each), kept in registers across all key blocks.
    H16 qf[2];
    {
        const _Float16* qp = &Qm[base + (size_t)(qb * 16 + lh) * rowstride];
#pragma unroll
        for (int kf = 0; kf < 2; ++kf) {
            const _Float16* p = qp + kf * 32 + hi * 8;
            qf[kf].h[0] = *(const h8*)p;
            qf[kf].h[1] = *(const h8*)(p + 16);
        }
    }

    f8 Oacc[4] = {};
    float mrun[8], lrun[8];
#pragma unroll
    for (int j = 0; j < 8; ++j) { mrun[j] = -3.0e38f; lrun[j] = 0.0f; }

    const int   qmax  = qb * 16 + 15;
    const int   nkb   = qmax / 32 + 1;
    const float scale = 0.125f;  // 1/sqrt(HD)

    for (int kb = 0; kb < nkb; ++kb) {
        const int k0 = kb * 32;

        // ---- stage V block (32 keys x 64 dh) into LDS; async overlaps softmax
        {
            const _Float16* vp = &Vm[base + (size_t)(k0 + lane) * rowstride];
            _Float16* dst = &lV[wv][lane * 64];
#pragma unroll
            for (int c = 0; c < 64; c += 8) {
#if HAVE_ASYNC_LDS
                async_copy_b128(vp + c, dst + c);
#else
                *(h8*)(dst + c) = *(const h8*)(vp + c);
#endif
            }
        }

        // ---- scores S[2]: 16 queries x 32 keys (two 16x16 tiles) ----
        f8 S[2] = {};
#pragma unroll
        for (int ns = 0; ns < 2; ++ns) {
            const _Float16* kp = &Km[base + (size_t)(k0 + ns * 16 + lh) * rowstride];
#pragma unroll
            for (int kf = 0; kf < 2; ++kf) {
                H16 bf;  // B-frag: lane = key col, halves = dh hi*8+[0..8) and hi*8+16+[0..8)
                const _Float16* p = kp + kf * 32 + hi * 8;
                bf.h[0] = *(const h8*)p;
                bf.h[1] = *(const h8*)(p + 16);
                S[ns] = __builtin_amdgcn_wmma_f32_16x16x32_f16(
                    false, qf[kf].v, false, bf.v, (short)0, S[ns], false, false);
            }
        }

        // ---- online softmax (per row j; each row lives on 16 lanes) ----
        float Pv[2][8], rmax[8], rsum[8];
#pragma unroll
        for (int j = 0; j < 8; ++j) {
            const int q = qb * 16 + j + hi * 8;
            float s0 = S[0][j] * scale + ((k0 + lh)      <= q ? 0.0f : -1.0e9f);
            float s1 = S[1][j] * scale + ((k0 + 16 + lh) <= q ? 0.0f : -1.0e9f);
            Pv[0][j] = s0; Pv[1][j] = s1;
            rmax[j]  = fmaxf(s0, s1);
        }
#pragma unroll
        for (int off = 1; off < 16; off <<= 1)
#pragma unroll
            for (int j = 0; j < 8; ++j)
                rmax[j] = fmaxf(rmax[j], __shfl_xor(rmax[j], off, 32));

#pragma unroll
        for (int j = 0; j < 8; ++j) {
            const float mnew  = fmaxf(mrun[j], rmax[j]);
            const float alpha = __expf(mrun[j] - mnew);
            mrun[j] = mnew;
            const float p0 = __expf(Pv[0][j] - mnew);
            const float p1 = __expf(Pv[1][j] - mnew);
            Pv[0][j] = p0; Pv[1][j] = p1;
            rsum[j]  = p0 + p1;
            lrun[j] *= alpha;
#pragma unroll
            for (int t = 0; t < 4; ++t) Oacc[t][j] *= alpha;
        }
#pragma unroll
        for (int off = 1; off < 16; off <<= 1)
#pragma unroll
            for (int j = 0; j < 8; ++j)
                rsum[j] += __shfl_xor(rsum[j], off, 32);
#pragma unroll
        for (int j = 0; j < 8; ++j) lrun[j] += rsum[j];

        // ---- transpose P (C layout -> A layout) through LDS ----
        {
            _Float16* pp = &lP[wv][0];
#pragma unroll
            for (int j = 0; j < 8; ++j) {
                const int m = j + hi * 8;
                pp[m * 32 + lh]      = (_Float16)Pv[0][j];
                pp[m * 32 + 16 + lh] = (_Float16)Pv[1][j];
            }
        }
        __builtin_amdgcn_wave_barrier();
        wait_asynccnt<0>();                             // V tile landed in LDS
        asm volatile("s_wait_dscnt 0" ::: "memory");    // P stores visible

        H16 pf;  // A-frag of P (16x32)
        {
            const _Float16* p = &lP[wv][lh * 32 + hi * 8];
            pf.h[0] = *(const h8*)p;
            pf.h[1] = *(const h8*)(p + 16);
        }

        // ---- O(16x64) += P(16x32) @ V(32x64) ----
#pragma unroll
        for (int t = 0; t < 4; ++t) {
            H16 vf;  // B-frag: lane = dh col t*16+lh, halves = keys hi*8+(i&7)+(i>>3)*16
            const _Float16* vb = &lV[wv][0];
#pragma unroll
            for (int i = 0; i < 16; ++i)
                vf.v[i] = vb[(hi * 8 + (i & 7) + (i >> 3) * 16) * 64 + t * 16 + lh];
            Oacc[t] = __builtin_amdgcn_wmma_f32_16x16x32_f16(
                false, pf.v, false, vf.v, (short)0, Oacc[t], false, false);
        }
        __builtin_amdgcn_wave_barrier();  // keep lV live until consumed
    }

    // ---- normalize and store attention output (f16, [B,S,DIM] layout) ----
#pragma unroll
    for (int t = 0; t < 4; ++t)
#pragma unroll
        for (int j = 0; j < 8; ++j) {
            const int q = qb * 16 + j + hi * 8;
            const float val = Oacc[t][j] / lrun[j];
            Om[base + (size_t)q * rowstride + t * 16 + lh] = (_Float16)val;
        }
}

// ---------------------------------------------------------------------------
// Host-side launch
// ---------------------------------------------------------------------------
extern "C" void kernel_launch(void* const* d_in, const int* in_sizes, int n_in,
                              void* d_out, int out_size, void* d_ws, size_t ws_size,
                              hipStream_t stream)
{
    const float* x  = (const float*)d_in[0];
    // d_in[1] = additive causal mask: applied analytically in-kernel, unused.
    const float* wq = (const float*)d_in[2];
    const float* wk = (const float*)d_in[3];
    const float* wv = (const float*)d_in[4];
    const float* wo = (const float*)d_in[5];
    float* out = (float*)d_out;

    // Workspace partition (f16): x, 4 weights, Q, K, V, attn-out.
    _Float16* ws  = (_Float16*)d_ws;
    _Float16* xh  = ws;
    _Float16* wqh = xh  + (size_t)MTOT * DIMD;
    _Float16* wkh = wqh + (size_t)DIMD * DIMD;
    _Float16* wvh = wkh + (size_t)DIMD * DIMD;
    _Float16* woh = wvh + (size_t)DIMD * DIMD;
    _Float16* qh  = woh + (size_t)DIMD * DIMD;
    _Float16* kh  = qh  + (size_t)MTOT * DIMD;
    _Float16* vh  = kh  + (size_t)MTOT * DIMD;
    _Float16* ah  = vh  + (size_t)MTOT * DIMD;

    const int nx = MTOT * DIMD;       // 4,194,304
    const int nw = DIMD * DIMD;       // 1,048,576
    cvt_f32_to_f16<<<nx / (256 * 4), 256, 0, stream>>>(x,  xh,  nx);
    cvt_f32_to_f16<<<nw / (256 * 4), 256, 0, stream>>>(wq, wqh, nw);
    cvt_f32_to_f16<<<nw / (256 * 4), 256, 0, stream>>>(wk, wkh, nw);
    cvt_f32_to_f16<<<nw / (256 * 4), 256, 0, stream>>>(wv, wvh, nw);
    cvt_f32_to_f16<<<nw / (256 * 4), 256, 0, stream>>>(wo, woh, nw);

    dim3 ggrid(DIMD / 128, MTOT / 128);   // (8, 32)
    gemm_xwt<_Float16><<<ggrid, 256, 0, stream>>>(xh, wqh, qh, MTOT, DIMD, DIMD);
    gemm_xwt<_Float16><<<ggrid, 256, 0, stream>>>(xh, wkh, kh, MTOT, DIMD, DIMD);
    gemm_xwt<_Float16><<<ggrid, 256, 0, stream>>>(xh, wvh, vh, MTOT, DIMD, DIMD);

    // 4096 wave-tasks (b,h,qblock), 4 waves per 128-thread block.
    flash_attn<<<(BATCH * NHEAD * (SEQ / 16)) / 4, 128, 0, stream>>>(qh, kh, vh, ah);

    gemm_xwt<float><<<ggrid, 256, 0, stream>>>(ah, woh, out, MTOT, DIMD, DIMD);
}
